// Attention_21878563405851
// MI455X (gfx1250) — matrix operand order repaired and checked by reading, not compiled, not run on hardware
//
#include <hip/hip_runtime.h>
#include <hip/hip_bf16.h>
#include <cstdint>

typedef __attribute__((ext_vector_type(16))) _Float16 v16h;
typedef __attribute__((ext_vector_type(8)))  _Float16 v8h;
typedef __attribute__((ext_vector_type(8)))  float    v8f;
typedef __attribute__((ext_vector_type(4)))  unsigned v4u;
typedef __attribute__((ext_vector_type(8)))  int      v8i;
typedef __attribute__((ext_vector_type(4)))  int      v4i;

union AF { v16h v; v8h h[2]; };

static __device__ __forceinline__ v8f wmma_f16(const AF& a, const AF& b, v8f c) {
    return __builtin_amdgcn_wmma_f32_16x16x32_f16(false, a.v, false, b.v,
                                                  (short)0, c, false, false);
}

#if __has_builtin(__builtin_amdgcn_sched_barrier)
#define SCHED_FENCE() __builtin_amdgcn_sched_barrier(0)
#else
#define SCHED_FENCE()
#endif

#define HAS_TDM __has_builtin(__builtin_amdgcn_tensor_load_to_lds)

#if HAS_TDM
// byte offset of a __shared__ object within LDS (generic -> AS(3) -> int)
static __device__ __forceinline__ unsigned lds_addr_of(const void* p) {
    return (unsigned)(uintptr_t)(__attribute__((address_space(3))) const void*)p;
}

// Issue one TDM descriptor: copy a 64x64 f16 tile (row stride = strideElems
// f16 elements) from global memory into LDS at byte offset ldsOff.
static __device__ __forceinline__ void tdm_load_tile64x64(const _Float16* gsrc,
                                                          unsigned ldsOff,
                                                          unsigned strideElems) {
    uint64_t ga = (uint64_t)(uintptr_t)gsrc;
    v4u g0;
    g0.x = 1u;                                              // count=1 (valid), user mode
    g0.y = ldsOff;                                          // lds_addr (bytes)
    g0.z = (unsigned)(ga & 0xFFFFFFFFu);                    // global_addr[31:0]
    g0.w = (unsigned)((ga >> 32) & 0x01FFFFFFu) | (2u << 30); // addr[56:32] | type=2
    v8i g1;
    g1[0] = 0x00010000;               // workgroup_mask=0, data_size=1 (2 bytes)
    g1[1] = (64 & 0xFFFF) << 16;      // tensor_dim0[15:0]=64 @bits[63:48]
    g1[2] = (64 & 0xFFFF) << 16;      // tensor_dim0[31:16]=0, tensor_dim1[15:0]=64
    g1[3] = (64 & 0xFFFF) << 16;      // tensor_dim1[31:16]=0, tile_dim0=64
    g1[4] = 64;                       // tile_dim1=64, tile_dim2=0
    g1[5] = (int)strideElems;         // tensor_dim0_stride[31:0]
    g1[6] = 0;                        // stride0[47:32]=0, dim1_stride[15:0]=0
    g1[7] = 0;                        // dim1_stride[47:16]=0
    v4i gz4 = {0, 0, 0, 0};           // groups 2/3 unused (2D tensor)
    v8i gz8 = {0, 0, 0, 0, 0, 0, 0, 0};
    __builtin_amdgcn_tensor_load_to_lds(g0, g1, gz4, gz4, gz8, 0);
}
#endif

// ---------------------------------------------------------------------------
// Generic 64x64-tile WMMA GEMM:  out[M,N] = A[M,K] * W[K,N]  (+ bias) * scale
// ---------------------------------------------------------------------------
template <bool A_F16, bool OUT_F16>
__global__ __launch_bounds__(128) void gemm_wmma(const void* __restrict__ Ap,
                                                 const float* __restrict__ W,
                                                 const float* __restrict__ bias,
                                                 void* __restrict__ Op,
                                                 int M, int Nout, int K,
                                                 float outScale) {
    __shared__ __align__(16) _Float16 As[64 * 32];   // [m][k]
    __shared__ __align__(16) _Float16 Ws[64 * 32];   // [n][k] (transposed)

    const int tid  = threadIdx.x;
    const int lane = tid & 31;
    const int wv   = tid >> 5;
    const int half = lane >> 4;
    const int l16  = lane & 15;
    const int n0 = blockIdx.x * 64;
    const int m0 = blockIdx.y * 64;

    const v8f vzero = {0.f, 0.f, 0.f, 0.f, 0.f, 0.f, 0.f, 0.f};
    v8f acc[4];
#pragma unroll
    for (int a = 0; a < 4; ++a) acc[a] = vzero;

    for (int k0 = 0; k0 < K; k0 += 32) {
        if (A_F16) {
            const _Float16* A = (const _Float16*)Ap;
#pragma unroll
            for (int i = 0; i < 2; ++i) {
                int e = tid + i * 128;
                int r = e >> 2, kc = (e & 3) * 8;
                *(v8h*)&As[r * 32 + kc] =
                    *(const v8h*)&A[(size_t)(m0 + r) * K + k0 + kc];
            }
        } else {
            const float* A = (const float*)Ap;
#pragma unroll
            for (int i = 0; i < 4; ++i) {
                int e = tid + i * 128;
                int r = e >> 3, kc = (e & 7) * 4;
                float4 v = *(const float4*)&A[(size_t)(m0 + r) * K + k0 + kc];
                const float* f = (const float*)&v;
#pragma unroll
                for (int j = 0; j < 4; ++j)
                    As[r * 32 + kc + j] = (_Float16)f[j];
            }
        }
#pragma unroll
        for (int i = 0; i < 4; ++i) {
            int e = tid + i * 128;
            int kr = e >> 4, nc = (e & 15) * 4;
            float4 v = *(const float4*)&W[(size_t)(k0 + kr) * Nout + n0 + nc];
            const float* f = (const float*)&v;
#pragma unroll
            for (int j = 0; j < 4; ++j)
                Ws[(nc + j) * 32 + kr] = (_Float16)f[j];
        }
        if (k0 + 32 < K)
            __builtin_prefetch(&W[(size_t)(k0 + 32 + (tid >> 4)) * Nout + n0], 0, 0);
        __syncthreads();

        // batch all fragment loads before the WMMA chain; sched fence stops
        // the scheduler from re-sinking loads next to their consumers
        AF af;
        const _Float16* abase = &As[(wv * 16 + l16) * 32];
        af.h[0] = *(const v8h*)(abase + 8 * half);
        af.h[1] = *(const v8h*)(abase + 16 + 8 * half);

        AF bf[4];
#pragma unroll
        for (int a = 0; a < 4; ++a) {
            const _Float16* bbase = &Ws[(a * 16 + l16) * 32];
            bf[a].h[0] = *(const v8h*)(bbase + 8 * half);
            bf[a].h[1] = *(const v8h*)(bbase + 16 + 8 * half);
        }
        SCHED_FENCE();
#pragma unroll
        for (int a = 0; a < 4; ++a)
            acc[a] = wmma_f16(af, bf[a], acc[a]);
        __syncthreads();
    }

#pragma unroll
    for (int a = 0; a < 4; ++a) {
#pragma unroll
        for (int r = 0; r < 8; ++r) {
            int row = m0 + wv * 16 + r + 8 * half;
            int col = n0 + a * 16 + l16;
            float val = acc[a][r] * outScale;
            if (OUT_F16) {
                ((_Float16*)Op)[(size_t)row * Nout + col] = (_Float16)val;
            } else {
                float bv = bias ? bias[col] : 0.f;
                ((float*)Op)[(size_t)row * Nout + col] = val + bv;
            }
        }
    }
}

// ---------------------------------------------------------------------------
// Flash-attention with TDM-double-buffered K/V tiles.
// grid = (N/64, B*H); 128 threads = 4 waves; 16 Q rows per wave.
// ---------------------------------------------------------------------------
__global__ __launch_bounds__(128) void attn_wmma(const _Float16* __restrict__ Qb,
                                                 const _Float16* __restrict__ KVb,
                                                 _Float16* __restrict__ Ob) {
    __shared__ __align__(16) _Float16 Ks[2][64 * 64];    // [kvrow][d]
    __shared__ __align__(16) _Float16 Vraw[2][64 * 64];  // [kvrow][d]
    __shared__ __align__(16) _Float16 Vt[64 * 64];       // [d][kvrow]
    __shared__ __align__(16) _Float16 Ps[4 * 16 * 64];   // per-wave P staging

    const int tid  = threadIdx.x;
    const int lane = tid & 31;
    const int wv   = tid >> 5;
    const int half = lane >> 4;
    const int l16  = lane & 15;
    const int bh = blockIdx.y;
    const int b  = bh >> 3;
    const int h  = bh & 7;
    const int q0 = blockIdx.x * 64;
    const size_t kvbase = (size_t)b * 2048;

    // --- this wave's Q fragment (16 rows x 64 d, two K-chunks) ---
    const size_t qrow = kvbase + q0 + wv * 16 + l16;
    AF qf[2];
#pragma unroll
    for (int c = 0; c < 2; ++c) {
        const _Float16* p = &Qb[qrow * 512 + h * 64 + c * 32];
        qf[c].h[0] = *(const v8h*)(p + 8 * half);
        qf[c].h[1] = *(const v8h*)(p + 16 + 8 * half);
    }

    const v8f vzero = {0.f, 0.f, 0.f, 0.f, 0.f, 0.f, 0.f, 0.f};
    v8f oacc[4];
#pragma unroll
    for (int a = 0; a < 4; ++a) oacc[a] = vzero;
    float mrow[8], lrow[8];
#pragma unroll
    for (int r = 0; r < 8; ++r) { mrow[r] = -3.0e38f; lrow[r] = 0.f; }

    _Float16* Pw = &Ps[wv * 16 * 64];
    const int ntiles = 2048 / 64;

    // ---- preload tile 0 into buffer 0 ----
#if HAS_TDM
    if (wv == 0) {
        const _Float16* g = &KVb[(kvbase + 0) * 1024 + h * 64];
        tdm_load_tile64x64(g,       lds_addr_of(&Ks[0][0]),   1024);
        tdm_load_tile64x64(g + 512, lds_addr_of(&Vraw[0][0]), 1024);
    }
#else
#pragma unroll
    for (int i = 0; i < 4; ++i) {
        int e = tid + i * 128;
        int r = e >> 3, c8 = (e & 7) * 8;
        size_t g = kvbase + r;
        *(v8h*)&Ks[0][r * 64 + c8]   = *(const v8h*)&KVb[g * 1024 + h * 64 + c8];
        *(v8h*)&Vraw[0][r * 64 + c8] = *(const v8h*)&KVb[g * 1024 + 512 + h * 64 + c8];
    }
#endif

    for (int it = 0; it < ntiles; ++it) {
        const int cur = it & 1;
        const _Float16* Kc = &Ks[cur][0];

        __syncthreads();   // prior iteration done reading the alt buffers
#if HAS_TDM
        if (wv == 0) {
            if (it + 1 < ntiles) {
                const _Float16* g = &KVb[(kvbase + (it + 1) * 64) * 1024 + h * 64];
                tdm_load_tile64x64(g,       lds_addr_of(&Ks[cur ^ 1][0]),   1024);
                tdm_load_tile64x64(g + 512, lds_addr_of(&Vraw[cur ^ 1][0]), 1024);
                // in-order TENSORcnt: <=2 outstanding => current tile's pair done
                __builtin_amdgcn_s_wait_tensorcnt(2);
            } else {
                __builtin_amdgcn_s_wait_tensorcnt(0);
            }
        }
#else
        if (it + 1 < ntiles) {
#pragma unroll
            for (int i = 0; i < 4; ++i) {
                int e = tid + i * 128;
                int r = e >> 3, c8 = (e & 7) * 8;
                size_t g = kvbase + (it + 1) * 64 + r;
                *(v8h*)&Ks[cur ^ 1][r * 64 + c8] =
                    *(const v8h*)&KVb[g * 1024 + h * 64 + c8];
                *(v8h*)&Vraw[cur ^ 1][r * 64 + c8] =
                    *(const v8h*)&KVb[g * 1024 + 512 + h * 64 + c8];
            }
        }
#endif
        __syncthreads();   // current tile's LDS data visible to all waves

        // ---- LDS->LDS transpose of V tile: Vraw[kv][d] -> Vt[d][kv] ----
#pragma unroll
        for (int i = 0; i < 4; ++i) {
            int e = tid + i * 128;
            int r = e >> 3, c8 = (e & 7) * 8;
            v8h v = *(const v8h*)&Vraw[cur][r * 64 + c8];
#pragma unroll
            for (int t = 0; t < 8; ++t) Vt[(c8 + t) * 64 + r] = v[t];
        }
        __syncthreads();

        // ---- S = Q * K^T : batch all 8 B-fragment loads, then 8 WMMAs ----
        AF kf[4][2];
#pragma unroll
        for (int a = 0; a < 4; ++a) {
            const _Float16* kb = &Kc[(a * 16 + l16) * 64];
#pragma unroll
            for (int c = 0; c < 2; ++c) {
                kf[a][c].h[0] = *(const v8h*)(kb + c * 32 + 8 * half);
                kf[a][c].h[1] = *(const v8h*)(kb + c * 32 + 16 + 8 * half);
            }
        }
        SCHED_FENCE();
        v8f sacc[4];
#pragma unroll
        for (int a = 0; a < 4; ++a) sacc[a] = vzero;
#pragma unroll
        for (int c = 0; c < 2; ++c)        // c outer: RAW chain distance 4
#pragma unroll
            for (int a = 0; a < 4; ++a)
                sacc[a] = wmma_f16(qf[c], kf[a][c], sacc[a]);

        // ---- online softmax (row m in the 16 lanes of half m/8; VGPR r -> row r+8*half)
        float mnew[8], rs[8];
#pragma unroll
        for (int r = 0; r < 8; ++r) {
            float mx = fmaxf(fmaxf(sacc[0][r], sacc[1][r]),
                             fmaxf(sacc[2][r], sacc[3][r]));
            mx = fmaxf(mx, __shfl_xor(mx, 1));
            mx = fmaxf(mx, __shfl_xor(mx, 2));
            mx = fmaxf(mx, __shfl_xor(mx, 4));
            mx = fmaxf(mx, __shfl_xor(mx, 8));
            mnew[r] = fmaxf(mrow[r], mx);
            rs[r] = 0.f;
        }
#pragma unroll
        for (int a = 0; a < 4; ++a) {
#pragma unroll
            for (int r = 0; r < 8; ++r) {
                float pv = __expf(sacc[a][r] - mnew[r]);
                rs[r] += pv;
                Pw[(r + 8 * half) * 64 + a * 16 + l16] = (_Float16)pv;
            }
        }
#pragma unroll
        for (int r = 0; r < 8; ++r) {
            rs[r] += __shfl_xor(rs[r], 1);
            rs[r] += __shfl_xor(rs[r], 2);
            rs[r] += __shfl_xor(rs[r], 4);
            rs[r] += __shfl_xor(rs[r], 8);
            float alpha = __expf(mrow[r] - mnew[r]);
            lrow[r] = lrow[r] * alpha + rs[r];
            mrow[r] = mnew[r];
#pragma unroll
            for (int a = 0; a < 4; ++a) oacc[a][r] *= alpha;
        }

        // ---- O += P * V : batch fragment loads, then 8 WMMAs ----
        AF pf[2];
        const _Float16* pb = &Pw[l16 * 64];
#pragma unroll
        for (int c = 0; c < 2; ++c) {
            pf[c].h[0] = *(const v8h*)(pb + c * 32 + 8 * half);
            pf[c].h[1] = *(const v8h*)(pb + c * 32 + 16 + 8 * half);
        }
        AF vf[4][2];
#pragma unroll
        for (int a = 0; a < 4; ++a) {
            const _Float16* vb = &Vt[(a * 16 + l16) * 64];
#pragma unroll
            for (int c = 0; c < 2; ++c) {
                vf[a][c].h[0] = *(const v8h*)(vb + c * 32 + 8 * half);
                vf[a][c].h[1] = *(const v8h*)(vb + c * 32 + 16 + 8 * half);
            }
        }
        SCHED_FENCE();
#pragma unroll
        for (int c = 0; c < 2; ++c)
#pragma unroll
            for (int a = 0; a < 4; ++a)
                oacc[a] = wmma_f16(pf[c], vf[a][c], oacc[a]);
    }

    // ---- normalize and write O ----
#pragma unroll
    for (int a = 0; a < 4; ++a) {
#pragma unroll
        for (int r = 0; r < 8; ++r) {
            size_t row = kvbase + q0 + wv * 16 + r + 8 * half;
            int col = h * 64 + a * 16 + l16;
            Ob[row * 512 + col] = (_Float16)(oacc[a][r] / lrow[r]);
        }
    }
}

// ---------------------------------------------------------------------------
extern "C" void kernel_launch(void* const* d_in, const int* in_sizes, int n_in,
                              void* d_out, int out_size, void* d_ws, size_t ws_size,
                              hipStream_t stream) {
    const float* x1   = (const float*)d_in[0];   // [2,2048,512]
    const float* x2   = (const float*)d_in[1];   // [2,2048,512]
    const float* Wqk  = (const float*)d_in[2];   // [512,1024]
    const float* Wv   = (const float*)d_in[3];   // [512,512]
    const float* Wout = (const float*)d_in[4];   // [512,512]
    const float* bout = (const float*)d_in[5];   // [512]
    float* out = (float*)d_out;                  // [2,2048,512]

    _Float16* KV = (_Float16*)d_ws;                  // [4096,1024] f16 (8 MB)
    _Float16* Q  = KV + (size_t)4096 * 1024;         // [4096,512]  f16 (4 MB)
    _Float16* O  = Q  + (size_t)4096 * 512;          // [4096,512]  f16 (4 MB)

    dim3 blk(128);
    gemm_wmma<false, true><<<dim3(1024 / 64, 4096 / 64), blk, 0, stream>>>(
        x1, Wqk, nullptr, KV, 4096, 1024, 512, 1.0f);
    gemm_wmma<false, true><<<dim3(512 / 64, 4096 / 64), blk, 0, stream>>>(
        x2, Wv, nullptr, Q, 4096, 512, 512, 0.125f);
    attn_wmma<<<dim3(2048 / 64, 16), blk, 0, stream>>>(Q, KV, O);
    gemm_wmma<true, false><<<dim3(512 / 64, 4096 / 64), blk, 0, stream>>>(
        O, Wout, bout, out, 4096, 512, 512, 1.0f);
}
